// DeepNormEncoderLayer_5471788335244
// MI455X (gfx1250) — compile-verified, matrix operand order checked by reading, not run on hardware
//
#include <hip/hip_runtime.h>
#include <hip/hip_bf16.h>
#include <math.h>

// ---------------------------------------------------------------------------
// Types for CDNA5 WMMA (gfx1250, wave32)
// ---------------------------------------------------------------------------
typedef __bf16 bf16_t;
typedef __attribute__((ext_vector_type(16))) __bf16 v16bf;
typedef __attribute__((ext_vector_type(8)))  float  v8f;

__device__ __forceinline__ v8f wmma_bf16(v16bf a, v16bf b, v8f c) {
  // v_wmma_f32_16x16x32_bf16: D = A(16x32) * B(32x16) + C(16x16 f32)
  return __builtin_amdgcn_wmma_f32_16x16x32_bf16(
      /*neg_a=*/false, a, /*neg_b=*/false, b,
      /*c_mod=*/(short)0, c, /*reuse_a=*/false, /*reuse_b=*/false);
}

// Load a 16x32 bf16 fragment (A-layout / our Bt-layout) from a row-major
// region with `stride` elements per row. Per ISA 7.12.2 (16-bit A 16x32):
//   lanes 0-15 : row = lane,     K = 0..7  (lo 16B) and 16..23 (hi 16B)
//   lanes 16-31: row = lane-16,  K = 8..15 (lo 16B) and 24..31 (hi 16B)
__device__ __forceinline__ v16bf load_frag(const bf16_t* base, int stride, int lane) {
  const int row = lane & 15;
  const int kb  = (lane < 16) ? 0 : 8;
  const bf16_t* p = base + (size_t)row * stride + kb;
  v16bf f;
  *((float4*)&f)       = *((const float4*)p);
  *(((float4*)&f) + 1) = *((const float4*)(p + 16));
  return f;
}

// CDNA5 async copy: GLOBAL_LOAD_ASYNC_TO_LDS_B128 (tracked by ASYNCcnt,
// completes in order -> perfect for double buffering). The low 32 bits of a
// generic pointer into LDS are the wave-relative DS byte offset (ISA §10.2).
__device__ __forceinline__ void async_load_b128(const bf16_t* gaddr, bf16_t* lptr) {
  unsigned loff = (unsigned)(size_t)lptr;
  asm volatile("global_load_async_to_lds_b128 %0, %1, off"
               :: "v"(loff), "v"(gaddr) : "memory");
}

// ---------------------------------------------------------------------------
// Elementwise converters
// ---------------------------------------------------------------------------
__global__ void cvt_f32_bf16_ker(const float* __restrict__ in,
                                 bf16_t* __restrict__ out, int n) {
  int i = blockIdx.x * blockDim.x + threadIdx.x;
  if (i < n) out[i] = (bf16_t)in[i];
}

// W [K,N] f32 (row-major) -> Wt [N,K] bf16 (row-major)
__global__ void wtrans_ker(const float* __restrict__ W, bf16_t* __restrict__ Wt,
                           int K, int N) {
  int i = blockIdx.x * blockDim.x + threadIdx.x;
  if (i < K * N) {
    int k = i / N, n = i - k * N;
    Wt[(size_t)n * K + k] = (bf16_t)W[i];
  }
}

// V bf16 [B*T, C] -> Vt bf16 [B][H][64][T]
__global__ void vtrans_ker(const bf16_t* __restrict__ V, bf16_t* __restrict__ Vt) {
  constexpr int T = 2048, C = 1024, H = 16;
  int i = blockIdx.x * blockDim.x + threadIdx.x;
  if (i < 2 * T * C) {
    int bt = i / C, c = i - bt * C;
    int b = bt / T, t = bt - b * T;
    int h = c >> 6, d = c & 63;
    Vt[(((size_t)(b * H + h)) * 64 + d) * T + t] = V[i];
  }
}

// ---------------------------------------------------------------------------
// Tiled bf16 GEMM:  C[M,N] = A[M,K] * Bt[N,K]^T  (fp32 accumulate)
// Block tile 128x128, 4 waves (2x2), 64x64 per wave (16 WMMA / K-step 32),
// double-buffered LDS filled by async-to-LDS copies (ASYNCcnt pipelined).
// EPI: 0 = store f32, 1 = store bf16, 2 = +bias, Mish, store bf16,
//      3 = +bias, store f32
// ---------------------------------------------------------------------------
template <int EPI>
__global__ __launch_bounds__(128) void gemm_bf16_ker(
    const bf16_t* __restrict__ A, const bf16_t* __restrict__ Bt,
    float* __restrict__ Cf, bf16_t* __restrict__ Cb,
    const float* __restrict__ bias, int M, int N, int K) {
  __shared__ bf16_t lA[2][128 * 40];  // 32 cols padded to 40 (bank spread)
  __shared__ bf16_t lB[2][128 * 40];

  const int tid  = threadIdx.x;        // 0..127
  const int lane = tid & 31;
  const int wid  = tid >> 5;           // 0..3
  const int wm   = wid & 1;            // 64-row half
  const int wn   = wid >> 1;           // 64-col half
  const int m0   = blockIdx.y * 128;
  const int n0   = blockIdx.x * 128;

  const int quad = tid & 3;            // 16B chunk within 32 cols
  const int lrow = tid >> 2;           // 0..31

  v8f acc[4][4] = {};

  // Issue one K-step tile (A 128x32 + B 128x32) = 8 async b128 per thread.
  auto issue = [&](int k0, int buf) {
    #pragma unroll
    for (int p = 0; p < 4; ++p) {
      const int row = lrow + p * 32;
      async_load_b128(A  + (size_t)(m0 + row) * K + k0 + quad * 8,
                      &lA[buf][row * 40 + quad * 8]);
      async_load_b128(Bt + (size_t)(n0 + row) * K + k0 + quad * 8,
                      &lB[buf][row * 40 + quad * 8]);
    }
  };

  issue(0, 0);
  int cur = 0;
  for (int k0 = 0; k0 < K; k0 += 32) {
    const bool has_next = (k0 + 32) < K;
    if (has_next) {
      issue(k0 + 32, cur ^ 1);
      // In-order completion: <=8 outstanding means previous tile landed.
      asm volatile("s_wait_asynccnt 0x8" ::: "memory");
    } else {
      asm volatile("s_wait_asynccnt 0x0" ::: "memory");
    }
    __syncthreads();

    v16bf af[4], bfr[4];
    #pragma unroll
    for (int i = 0; i < 4; ++i)
      af[i] = load_frag(&lA[cur][(wm * 64 + i * 16) * 40], 40, lane);
    #pragma unroll
    for (int j = 0; j < 4; ++j)
      bfr[j] = load_frag(&lB[cur][(wn * 64 + j * 16) * 40], 40, lane);
    #pragma unroll
    for (int i = 0; i < 4; ++i)
      #pragma unroll
      for (int j = 0; j < 4; ++j)
        acc[i][j] = wmma_bf16(af[i], bfr[j], acc[i][j]);

    __syncthreads();  // all waves done reading `cur` before it is refilled
    cur ^= 1;
  }

  // Epilogue. C layout (16x16 f32): VGPR r -> row r (lanes 0-15) / r+8
  // (lanes 16-31); col = lane&15.
  const int rbase = m0 + wm * 64 + ((lane < 16) ? 0 : 8);
  const int cbase = n0 + wn * 64 + (lane & 15);
  #pragma unroll
  for (int i = 0; i < 4; ++i) {
    #pragma unroll
    for (int j = 0; j < 4; ++j) {
      const int col = cbase + j * 16;
      float bv = 0.0f;
      if (EPI == 2 || EPI == 3) bv = bias[col];
      #pragma unroll
      for (int r = 0; r < 8; ++r) {
        const int row = rbase + i * 16 + r;
        float v = acc[i][j][r];
        if (EPI == 2 || EPI == 3) v += bv;
        if (EPI == 2) {  // Mish: v * tanh(softplus(v))
          float sp = (v > 8.0f) ? v : log1pf(__expf(v));
          v = v * tanhf(sp);
        }
        if (EPI == 0 || EPI == 3) Cf[(size_t)row * N + col] = v;
        else                      Cb[(size_t)row * N + col] = (bf16_t)v;
      }
    }
  }
}

// ---------------------------------------------------------------------------
// Flash attention: each wave owns 16 queries of one (b,h); streams keys in
// chunks of 32. Q[b,t,h*64+d], K[b,t,h*64+d] bf16; Vt[b,h,d,t] bf16.
// ---------------------------------------------------------------------------
__global__ __launch_bounds__(256) void attn_ker(const bf16_t* __restrict__ Qb,
                                                const bf16_t* __restrict__ Kb,
                                                const bf16_t* __restrict__ Vt,
                                                bf16_t* __restrict__ Ctx) {
  constexpr int T = 2048, C = 1024, H = 16;
  constexpr float scale = 0.125f;  // 1/sqrt(64)
  __shared__ bf16_t pls[8][16 * 40];  // per-wave P staging (16x32, padded)

  const int lane = threadIdx.x & 31;
  const int wid  = threadIdx.x >> 5;
  const int g    = blockIdx.x * 8 + wid;  // 0..4095
  const int QT   = T / 16;                // 128 query tiles
  const int qt   = g & (QT - 1);
  const int bh   = g / QT;                // 0..31
  const int h    = bh & (H - 1);
  const int b    = bh >> 4;

  const bf16_t* qbase = Qb + ((size_t)(b * T + qt * 16)) * C + h * 64;
  const v16bf qa0 = load_frag(qbase, C, lane);       // d = 0..31
  const v16bf qa1 = load_frag(qbase + 32, C, lane);  // d = 32..63

  float mrow[8], lrow[8];
  v8f cacc[4] = {};
  for (int r = 0; r < 8; ++r) { mrow[r] = -1e30f; lrow[r] = 0.0f; }

  const bf16_t* kbase = Kb + ((size_t)(b * T)) * C + h * 64;
  const bf16_t* vbase = Vt + ((size_t)bh) * 64 * T;
  bf16_t* pw = &pls[wid][0];

  for (int j = 0; j < T; j += 32) {
    const bf16_t* kp = kbase + (size_t)j * C;
    v16bf k00 = load_frag(kp, C, lane);
    v16bf k01 = load_frag(kp + 32, C, lane);
    v16bf k10 = load_frag(kp + (size_t)16 * C, C, lane);
    v16bf k11 = load_frag(kp + (size_t)16 * C + 32, C, lane);

    v8f s0 = {}, s1 = {};
    s0 = wmma_bf16(qa0, k00, s0);  s0 = wmma_bf16(qa1, k01, s0);
    s1 = wmma_bf16(qa0, k10, s1);  s1 = wmma_bf16(qa1, k11, s1);

    for (int r = 0; r < 8; ++r) {
      float a = s0[r] * scale, c2 = s1[r] * scale;
      // Row max across the 16 lanes holding this row's columns.
      float mx = fmaxf(a, c2);
      mx = fmaxf(mx, __shfl_xor(mx, 1));
      mx = fmaxf(mx, __shfl_xor(mx, 2));
      mx = fmaxf(mx, __shfl_xor(mx, 4));
      mx = fmaxf(mx, __shfl_xor(mx, 8));
      float mn   = fmaxf(mrow[r], mx);
      float corr = __expf(mrow[r] - mn);
      mrow[r] = mn;
      float p0 = __expf(a - mn), p1 = __expf(c2 - mn);
      float rs = p0 + p1;
      rs += __shfl_xor(rs, 1);
      rs += __shfl_xor(rs, 2);
      rs += __shfl_xor(rs, 4);
      rs += __shfl_xor(rs, 8);
      lrow[r] = lrow[r] * corr + rs;
      for (int dt = 0; dt < 4; ++dt) cacc[dt][r] *= corr;
      // Stage P (row-major 16x32) for reload in A-fragment layout.
      const int rr = r + ((lane < 16) ? 0 : 8);
      pw[rr * 40 + (lane & 15)]      = (bf16_t)p0;
      pw[rr * 40 + 16 + (lane & 15)] = (bf16_t)p1;
    }
    asm volatile("s_wait_dscnt 0" ::: "memory");

    v16bf pa = load_frag(pw, 40, lane);
    for (int dt = 0; dt < 4; ++dt) {
      const bf16_t* vp = vbase + (size_t)(dt * 16) * T + j;
      v16bf vf = load_frag(vp, T, lane);
      cacc[dt] = wmma_bf16(pa, vf, cacc[dt]);
    }
  }

  // Normalize and write ctx back in [B*T, C] bf16 layout.
  const int rhalf = (lane < 16) ? 0 : 8;
  for (int r = 0; r < 8; ++r) {
    float inv = 1.0f / lrow[r];
    int row = qt * 16 + r + rhalf;
    bf16_t* out = Ctx + ((size_t)(b * T + row)) * C + h * 64 + (lane & 15);
    for (int dt = 0; dt < 4; ++dt) out[dt * 16] = (bf16_t)(cacc[dt][r] * inv);
  }
}

// ---------------------------------------------------------------------------
// Fused DeepNorm residual + LayerNorm: y = LN(alpha*X + R)*g + be
// One block per 1024-element row; writes f32 and optional bf16 copy.
// ---------------------------------------------------------------------------
__global__ __launch_bounds__(256) void ln_ker(const float* __restrict__ X,
                                              const float* __restrict__ R,
                                              const float* __restrict__ g,
                                              const float* __restrict__ be,
                                              float* __restrict__ outF,
                                              bf16_t* __restrict__ outB,
                                              float alpha) {
  constexpr int C = 1024;
  __shared__ float red[16];
  const int row = blockIdx.x;
  const int tid = threadIdx.x;
  float v[4], s = 0.0f, s2 = 0.0f;
  for (int i = 0; i < 4; ++i) {
    int c = tid + i * 256;
    float val = alpha * X[(size_t)row * C + c] + R[(size_t)row * C + c];
    v[i] = val; s += val; s2 += val * val;
  }
  for (int d = 1; d < 32; d <<= 1) { s += __shfl_xor(s, d); s2 += __shfl_xor(s2, d); }
  const int lane = tid & 31, wid = tid >> 5;
  if (lane == 0) { red[wid] = s; red[8 + wid] = s2; }
  __syncthreads();
  if (wid == 0) {
    float a  = (lane < 8) ? red[lane] : 0.0f;
    float b2 = (lane < 8) ? red[8 + lane] : 0.0f;
    for (int d = 1; d < 8; d <<= 1) { a += __shfl_xor(a, d); b2 += __shfl_xor(b2, d); }
    if (lane == 0) { red[0] = a; red[1] = b2; }
  }
  __syncthreads();
  const float mu   = red[0] / C;
  const float var  = red[1] / C - mu * mu;
  const float rstd = rsqrtf(var + 1e-5f);
  for (int i = 0; i < 4; ++i) {
    int c = tid + i * 256;
    float y = (v[i] - mu) * rstd * g[c] + be[c];
    if (outF) outF[(size_t)row * C + c] = y;
    if (outB) outB[(size_t)row * C + c] = (bf16_t)y;
  }
}

// ---------------------------------------------------------------------------
// Host orchestration
// ---------------------------------------------------------------------------
extern "C" void kernel_launch(void* const* d_in, const int* in_sizes, int n_in,
                              void* d_out, int out_size, void* d_ws, size_t ws_size,
                              hipStream_t stream) {
  constexpr int B = 2, T = 2048, C = 1024, F = 4096;
  constexpr int M = B * T;  // 4096 rows
  const float* x   = (const float*)d_in[0];
  const float* Wq  = (const float*)d_in[1];
  const float* Wk  = (const float*)d_in[2];
  const float* Wv  = (const float*)d_in[3];
  const float* Wo  = (const float*)d_in[4];
  const float* W1  = (const float*)d_in[5];
  const float* b1  = (const float*)d_in[6];
  const float* W2  = (const float*)d_in[7];
  const float* b2  = (const float*)d_in[8];
  const float* g1  = (const float*)d_in[9];
  const float* be1 = (const float*)d_in[10];
  const float* g2  = (const float*)d_in[11];
  const float* be2 = (const float*)d_in[12];

  constexpr size_t MB = 1024ull * 1024ull;
  char* ws = (char*)d_ws;
  bf16_t* xb   = (bf16_t*)(ws + 0 * MB);    // 8 MB  [M,C] bf16
  bf16_t* Wqt  = (bf16_t*)(ws + 8 * MB);    // 2 MB  [C,C]^T
  bf16_t* Wkt  = (bf16_t*)(ws + 10 * MB);
  bf16_t* Wvt  = (bf16_t*)(ws + 12 * MB);
  bf16_t* Wot  = (bf16_t*)(ws + 14 * MB);
  bf16_t* W1t  = (bf16_t*)(ws + 16 * MB);   // 8 MB  [F,C]
  bf16_t* W2t  = (bf16_t*)(ws + 24 * MB);   // 8 MB  [C,F]
  bf16_t* Qb   = (bf16_t*)(ws + 32 * MB);   // 8 MB
  bf16_t* Kbb  = (bf16_t*)(ws + 40 * MB);   // 8 MB
  bf16_t* Vb   = (bf16_t*)(ws + 48 * MB);   // 8 MB
  bf16_t* Vtr  = (bf16_t*)(ws + 56 * MB);   // 8 MB
  bf16_t* hb   = (bf16_t*)(ws + 32 * MB);   // 32 MB, reuses dead Q/K/V/Vt
  bf16_t* ctxb = (bf16_t*)(ws + 64 * MB);   // 8 MB
  float*  attn = (float*)(ws + 72 * MB);    // 16 MB, reused as FF output
  float*  fff  = (float*)(ws + 72 * MB);
  float*  x1f  = (float*)(ws + 88 * MB);    // 16 MB
  bf16_t* x1b  = (bf16_t*)(ws + 104 * MB);  // 8 MB  -> total 112 MB
  (void)ws_size; (void)in_sizes; (void)n_in; (void)out_size;

  const int nXC = M * C;      // 4,194,304
  const int nCC = C * C;      // 1,048,576
  const int nCF = C * F;      // 4,194,304

  // 1) Precision conversion / weight transposes.
  cvt_f32_bf16_ker<<<(nXC + 255) / 256, 256, 0, stream>>>(x, xb, nXC);
  wtrans_ker<<<(nCC + 255) / 256, 256, 0, stream>>>(Wq, Wqt, C, C);
  wtrans_ker<<<(nCC + 255) / 256, 256, 0, stream>>>(Wk, Wkt, C, C);
  wtrans_ker<<<(nCC + 255) / 256, 256, 0, stream>>>(Wv, Wvt, C, C);
  wtrans_ker<<<(nCC + 255) / 256, 256, 0, stream>>>(Wo, Wot, C, C);
  wtrans_ker<<<(nCF + 255) / 256, 256, 0, stream>>>(W1, W1t, C, F);
  wtrans_ker<<<(nCF + 255) / 256, 256, 0, stream>>>(W2, W2t, F, C);

  // 2) QKV projections (WMMA GEMMs, bf16 out). Block tile 128x128, 128 thr.
  dim3 gP(C / 128, M / 128);
  gemm_bf16_ker<1><<<gP, 128, 0, stream>>>(xb, Wqt, nullptr, Qb,  nullptr, M, C, C);
  gemm_bf16_ker<1><<<gP, 128, 0, stream>>>(xb, Wkt, nullptr, Kbb, nullptr, M, C, C);
  gemm_bf16_ker<1><<<gP, 128, 0, stream>>>(xb, Wvt, nullptr, Vb,  nullptr, M, C, C);
  vtrans_ker<<<(nXC + 255) / 256, 256, 0, stream>>>(Vb, Vtr);

  // 3) Flash attention (WMMA): 4096 wave-tiles / 8 waves per block.
  attn_ker<<<512, 256, 0, stream>>>(Qb, Kbb, Vtr, ctxb);

  // 4) Output projection + DeepNorm LN1.
  gemm_bf16_ker<0><<<gP, 128, 0, stream>>>(ctxb, Wot, attn, nullptr, nullptr, M, C, C);
  ln_ker<<<M, 256, 0, stream>>>(x, attn, g1, be1, x1f, x1b, 2.0f);

  // 5) FFN: GEMM+bias+Mish -> bf16, then GEMM+bias -> f32.
  dim3 gF1(F / 128, M / 128);
  gemm_bf16_ker<2><<<gF1, 128, 0, stream>>>(x1b, W1t, nullptr, hb, b1, M, F, C);
  gemm_bf16_ker<3><<<gP, 128, 0, stream>>>(hb, W2t, fff, nullptr, b2, M, C, F);

  // 6) DeepNorm LN2 -> d_out (fp32).
  ln_ker<<<M, 256, 0, stream>>>(x1f, fff, g2, be2, (float*)d_out, nullptr, 2.0f);
}
